// Network_38242388804071
// MI455X (gfx1250) — matrix operand order repaired and checked by reading, not compiled, not run on hardware
//
#include <hip/hip_runtime.h>
#include <hip/hip_bf16.h>

typedef __attribute__((ext_vector_type(16))) _Float16 v16h;
typedef __attribute__((ext_vector_type(8)))  float    v8f;

#define WMMA_F16(a, b, c) \
    __builtin_amdgcn_wmma_f32_16x16x32_f16(false, (a), false, (b), (short)0, (c), false, false)

// ---------------------------------------------------------------------------
// WMMA fragment helpers (wave32, A/B in contiguous k order k = 16*(lane/16) + s)
// ---------------------------------------------------------------------------

// A matrix 16x32 f16: lanes 0-15 -> row M=lane, K = k0..k0+15
//                     lanes 16-31 -> row M=lane-16, K = k0+16..k0+31
__device__ __forceinline__ v16h load_a_frag(const _Float16* H, int pitch, int m0, int k0, int lane) {
    int m  = m0 + (lane & 15);
    int kb = k0 + ((lane >> 4) << 4);
    const _Float16* p = H + m * pitch + kb;
    v16h a;
#pragma unroll
    for (int j = 0; j < 8; ++j) a[j] = p[j];
#pragma unroll
    for (int j = 0; j < 8; ++j) a[j + 8] = p[8 + j];
    return a;
}

// B matrix 32x16 f16 where B[k][n] = W[n][k], W stored row-major [Cout][K] in LDS.
// lanes 0-15 -> col n=lane, K = k0..k0+15; lanes 16-31 -> col n=lane-16, K = k0+16..k0+31
__device__ __forceinline__ v16h load_b_frag(const _Float16* W, int pitch, int n0, int k0, int lane) {
    int n  = n0 + (lane & 15);
    int kb = k0 + ((lane >> 4) << 4);
    const _Float16* p = W + n * pitch + kb;
    v16h b;
#pragma unroll
    for (int j = 0; j < 16; ++j) b[j] = p[j];
    return b;
}

// C/D 16x16 f32: VGPR r holds (M = r + 8*(lane>=16), N = lane&15)
__device__ __forceinline__ v8f bias_frag(const float* bs, int n0, int lane) {
    float bv = bs[n0 + (lane & 15)];
    v8f c;
#pragma unroll
    for (int r = 0; r < 8; ++r) c[r] = bv;
    return c;
}

__device__ __forceinline__ void store_relu_frag(_Float16* H, int pitch, int m0, int n0, int lane, v8f d) {
    int hi = lane >> 4, nn = lane & 15;
#pragma unroll
    for (int r = 0; r < 8; ++r)
        H[(m0 + r + 8 * hi) * pitch + n0 + nn] = (_Float16)fmaxf(d[r], 0.0f);
}

// One 64-wide layer: O[64][64] = relu(X[64][K] @ W^T + b), W in LDS [64][K] (pitch==K padded)
__device__ __forceinline__ void gemm_layer64(const _Float16* X, int pitch, int K,
                                             const _Float16* W, const float* bs,
                                             _Float16* O, int rw, int chalf, int lane) {
    int m0 = rw * 16;
#pragma unroll
    for (int ct = 0; ct < 2; ++ct) {
        int n0 = chalf * 32 + ct * 16;
        v8f c = bias_frag(bs, n0, lane);
        for (int k0 = 0; k0 < K; k0 += 32) {
            v16h a = load_a_frag(X, pitch, m0, k0, lane);
            v16h b = load_b_frag(W, pitch, n0, k0, lane);
            c = WMMA_F16(a, b, c);
        }
        store_relu_frag(O, 64, m0, n0, lane, c);
    }
}

// ---------------------------------------------------------------------------
// Kernel A: encoder/decoder (tiny, scalar). One block (64 thr) per node.
// ---------------------------------------------------------------------------
__global__ __launch_bounds__(64) void encdec_kernel(
    const float* __restrict__ poses, const float* __restrict__ t_ped, const float* __restrict__ w,
    const float* __restrict__ Wec1,  const float* __restrict__ bec1,
    const float* __restrict__ Wec21, const float* __restrict__ bec21,
    const float* __restrict__ Wec22, const float* __restrict__ bec22,
    const float* __restrict__ Wdc1,  const float* __restrict__ bdc1,
    const float* __restrict__ Wdc21, const float* __restrict__ bdc21,
    const float* __restrict__ Wdc22, const float* __restrict__ bdc22,
    float* __restrict__ o_ei, float* __restrict__ o_dn,
    float* __restrict__ o_mean, float* __restrict__ o_logv)
{
    int n = blockIdx.x, tid = threadIdx.x;
    __shared__ float xin[88];
    __shared__ float net[64];
    __shared__ float zb[8];

    for (int c = tid; c < 88; c += 64) {
        float v;
        if (c < 72) { int k = c / 3, a = c - 3 * k; v = w[n * 24 + k] * poses[(n * 24 + k) * 3 + a]; }
        else v = t_ped[c - 72];
        xin[c] = v;
    }
    __syncthreads();
    {
        const float* Wr = Wec1 + ((size_t)n * 64 + tid) * 88;
        float s = bec1[n * 64 + tid];
        for (int c = 0; c < 88; ++c) s += Wr[c] * xin[c];
        net[tid] = fmaxf(s, 0.0f);
    }
    __syncthreads();
    if (tid < 8) {
        const float* Wr = Wec21 + ((size_t)n * 8 + tid) * 64;
        float s = bec21[n * 8 + tid];
        for (int i = 0; i < 64; ++i) s += Wr[i] * net[i];
        o_mean[n * 8 + tid] = s;
        zb[tid] = s;
        const float* Wr2 = Wec22 + ((size_t)n * 8 + tid) * 64;
        float s2 = bec22[n * 8 + tid];
        for (int i = 0; i < 64; ++i) s2 += Wr2[i] * net[i];
        o_logv[n * 8 + tid] = s2;
    }
    __syncthreads();
    if (tid < 8) xin[72 + tid] = zb[tid];   // dec_in = [poses_w(72), z(8)]
    __syncthreads();
    {
        const float* Wr = Wdc1 + ((size_t)n * 64 + tid) * 80;
        float s = bdc1[n * 64 + tid];
        for (int c = 0; c < 80; ++c) s += Wr[c] * xin[c];
        net[tid] = fmaxf(s, 0.0f);
    }
    __syncthreads();
    if (tid < 32) {
        const float* Wr = Wdc21 + ((size_t)n * 32 + tid) * 64;
        float s = bdc21[n * 32 + tid];
        for (int i = 0; i < 64; ++i) s += Wr[i] * net[i];
        o_ei[n * 32 + tid] = s;
    }
    if (tid < 3) {
        const float* Wr = Wdc22 + ((size_t)n * 3 + tid) * 64;
        float s = bdc22[n * 3 + tid];
        for (int i = 0; i < 64; ++i) s += Wr[i] * net[i];
        o_dn[n * 3 + tid] = s;
    }
}

// ---------------------------------------------------------------------------
// Kernel B: RBF blend weights. wgt[n][v] = mask * (bweights+1e-7)/sum
// ---------------------------------------------------------------------------
__global__ __launch_bounds__(256) void blendw_kernel(
    const float* __restrict__ wpts, const float* __restrict__ nodes, float* __restrict__ wgt)
{
    int v = blockIdx.x * 256 + threadIdx.x;
    if (v >= 2048) return;
    float px = wpts[v * 3 + 0], py = wpts[v * 3 + 1], pz = wpts[v * 3 + 2];
    const float inv2s2 = 1.0f / (2.0f * 0.3f * 0.3f);
    float sum = 0.0f;
    for (int n = 0; n < 128; ++n) {
        float dx = px - nodes[n * 3 + 0], dy = py - nodes[n * 3 + 1], dz = pz - nodes[n * 3 + 2];
        float d2 = dx * dx + dy * dy + dz * dz;
        float influ = expf(-d2 * inv2s2) - 0.01f;
        sum += fmaxf(influ, 0.0f) + 1e-7f;
    }
    float inv = 1.0f / sum;
    for (int n = 0; n < 128; ++n) {
        float dx = px - nodes[n * 3 + 0], dy = py - nodes[n * 3 + 1], dz = pz - nodes[n * 3 + 2];
        float d2 = dx * dx + dy * dy + dz * dz;
        float influ = expf(-d2 * inv2s2) - 0.01f;
        wgt[n * 2048 + v] = (influ >= 0.0f) ? (influ + 1e-7f) * inv : 0.0f;
    }
}

// ---------------------------------------------------------------------------
// Kernel C: feature-field 4-layer MLP (WMMA f16) + fused node-blend.
// Grid: (V/64, G). Block: 256 thr = 8 waves; wave = (row-tile rw, col-half).
// Each block loops over 128/G nodes, accumulating wgt-weighted relu(f4) in regs,
// then writes one deterministic partial sum [g][2048][256].
// global_prefetch (speculative, WGP-visible) pulls upcoming layer weights
// toward the WGP while the current layer computes.
// ---------------------------------------------------------------------------
__global__ __launch_bounds__(256) void feat_blend_kernel(
    const float* __restrict__ ei,   // [128][32]
    const float* __restrict__ lc,   // [128][2048][63]
    const float* __restrict__ Wf1, const float* __restrict__ bf1,
    const float* __restrict__ Wf2, const float* __restrict__ bf2,
    const float* __restrict__ Wf3, const float* __restrict__ bf3,
    const float* __restrict__ Wf4, const float* __restrict__ bf4,
    const float* __restrict__ wgt,  // [128][2048]
    float* __restrict__ partial,    // [G][2048][256]
    int nodes_per_group)
{
    __shared__ __align__(16) _Float16 Xs[64 * 96];   // layer-1 input (95 -> pad 96)
    __shared__ __align__(16) _Float16 Ha[64 * 64];
    __shared__ __align__(16) _Float16 Hb[64 * 64];
    __shared__ __align__(16) _Float16 Ws[256 * 64];  // reused weight buffer (max W4)
    __shared__ float bs[256];
    __shared__ float wgs[64];

    int tid   = threadIdx.x;
    int lane  = tid & 31;
    int wave  = tid >> 5;
    int rw    = wave & 3;      // row tile (16 rows each)
    int chalf = wave >> 2;     // column half
    int v0    = blockIdx.x * 64;
    int g     = blockIdx.y;

    v8f acc[8];
#pragma unroll
    for (int i = 0; i < 8; ++i) { v8f z = {0, 0, 0, 0, 0, 0, 0, 0}; acc[i] = z; }

    for (int t = 0; t < nodes_per_group; ++t) {
        int n = g * nodes_per_group + t;

        // Speculative prefetch of this node's later-layer weights (one cacheline
        // per 256B stride per thread); no counters involved (FLAT/GLOBAL_PREFETCH_B8).
        {
            const char* p2 = (const char*)(Wf2 + (size_t)n * 4096);
            const char* p3 = (const char*)(Wf3 + (size_t)n * 4096);
            const char* p4 = (const char*)(Wf4 + (size_t)n * 16384);
            int o = tid * 256;
            if (o < 16384) { __builtin_prefetch(p2 + o, 0, 1); __builtin_prefetch(p3 + o, 0, 1); }
            if (o < 65536) { __builtin_prefetch(p4 + o, 0, 1); }
        }

        // Stage layer-1 input [ei(32) | local_coords(63) | 0] and W1 (pad 95->96)
        for (int idx = tid; idx < 64 * 96; idx += 256) {
            int m = idx / 96, c = idx - m * 96;
            float v;
            if (c < 32)      v = ei[n * 32 + c];
            else if (c < 95) v = lc[((size_t)n * 2048 + (v0 + m)) * 63 + (c - 32)];
            else             v = 0.0f;
            Xs[idx] = (_Float16)v;
        }
        if (tid < 64) wgs[tid] = wgt[n * 2048 + v0 + tid];
        for (int idx = tid; idx < 64 * 96; idx += 256) {
            int i = idx / 96, c = idx - i * 96;
            Ws[idx] = (_Float16)((c < 95) ? Wf1[((size_t)n * 64 + i) * 95 + c] : 0.0f);
        }
        if (tid < 64) bs[tid] = bf1[n * 64 + tid];
        __syncthreads();
        gemm_layer64(Xs, 96, 96, Ws, bs, Ha, rw, chalf, lane);     // layer 1
        __syncthreads();

        for (int idx = tid; idx < 64 * 64; idx += 256) Ws[idx] = (_Float16)Wf2[(size_t)n * 4096 + idx];
        if (tid < 64) bs[tid] = bf2[n * 64 + tid];
        __syncthreads();
        gemm_layer64(Ha, 64, 64, Ws, bs, Hb, rw, chalf, lane);     // layer 2
        __syncthreads();

        for (int idx = tid; idx < 64 * 64; idx += 256) Ws[idx] = (_Float16)Wf3[(size_t)n * 4096 + idx];
        if (tid < 64) bs[tid] = bf3[n * 64 + tid];
        __syncthreads();
        gemm_layer64(Hb, 64, 64, Ws, bs, Ha, rw, chalf, lane);     // layer 3
        __syncthreads();

        for (int idx = tid; idx < 256 * 64; idx += 256) Ws[idx] = (_Float16)Wf4[(size_t)n * 16384 + idx];
        bs[tid] = bf4[n * 256 + tid];

        // Prefetch next node's W1 + layer-1 inputs while layer 4 runs.
        if (t + 1 < nodes_per_group) {
            int n2 = n + 1;
            const char* p1 = (const char*)(Wf1 + (size_t)n2 * 64 * 95);
            const char* pl = (const char*)(lc + ((size_t)n2 * 2048 + v0) * 63);
            int o = tid * 256;
            if (o < 24320) { __builtin_prefetch(p1 + o, 0, 1); }
            if (o < 16128) { __builtin_prefetch(pl + o, 0, 1); }   // 64*63*4 bytes
        }
        __syncthreads();

        // layer 4 (64 -> 256) + relu + wgt-weighted accumulate into registers
        int hi = lane >> 4;
        float wv[8];
#pragma unroll
        for (int r = 0; r < 8; ++r) wv[r] = wgs[rw * 16 + r + 8 * hi];
        int m0 = rw * 16;
#pragma unroll
        for (int ct = 0; ct < 8; ++ct) {
            int n0 = chalf * 128 + ct * 16;
            v8f c = bias_frag(bs, n0, lane);
#pragma unroll
            for (int k0 = 0; k0 < 64; k0 += 32) {
                v16h a = load_a_frag(Ha, 64, m0, k0, lane);
                v16h b = load_b_frag(Ws, 64, n0, k0, lane);
                c = WMMA_F16(a, b, c);
            }
#pragma unroll
            for (int r = 0; r < 8; ++r) acc[ct][r] += wv[r] * fmaxf(c[r], 0.0f);
        }
        __syncthreads();
    }

    // write partial blend sums (f32)
    size_t base = (size_t)g * 2048 * 256;
    int hi = lane >> 4, nn = lane & 15;
#pragma unroll
    for (int ct = 0; ct < 8; ++ct) {
        int c = chalf * 128 + ct * 16 + nn;
#pragma unroll
        for (int r = 0; r < 8; ++r) {
            int v = v0 + rw * 16 + r + 8 * hi;
            partial[base + (size_t)v * 256 + c] = acc[ct][r];
        }
    }
}

// ---------------------------------------------------------------------------
// Kernel D: reduce partials -> f_blend, NeRF head via WMMA.
// B matrix = [Wc1 (64 cols) | Wd1 (1 col) | zero-pad] = 80 cols. Dynamic LDS.
// ---------------------------------------------------------------------------
__global__ __launch_bounds__(256) void head_kernel(
    const float* __restrict__ partial, int G,
    const float* __restrict__ Wc1, const float* __restrict__ bc1,
    const float* __restrict__ Wc2, const float* __restrict__ bc2,
    const float* __restrict__ Wd1, const float* __restrict__ bd1,
    float* __restrict__ out_c, float* __restrict__ out_d)
{
    extern __shared__ __align__(16) char smem[];
    _Float16* Fs = (_Float16*)smem;                    // [128][256] f16   65536 B
    _Float16* Wh = (_Float16*)(smem + 65536);          // [80][256] f16    40960 B
    _Float16* Hn = (_Float16*)(smem + 106496);         // [128][64] f16    16384 B
    float*    bh = (float*)(smem + 122880);            // [80] f32

    int tid = threadIdx.x, lane = tid & 31, wave = tid >> 5;
    int v0 = blockIdx.x * 128;

    // deterministic reduction of node-group partials -> f_blend (f16)
    for (int idx = tid; idx < 128 * 256; idx += 256) {
        int vv = idx >> 8, c = idx & 255;
        float s = 0.0f;
        for (int g = 0; g < G; ++g)
            s += partial[(size_t)g * 2048 * 256 + (size_t)(v0 + vv) * 256 + c];
        Fs[idx] = (_Float16)s;
    }
    // head weights, transposed to [out][K] layout for load_b_frag
    for (int idx = tid; idx < 80 * 256; idx += 256) {
        int i = idx >> 8, c = idx & 255;
        float v = (i < 64) ? Wc1[c * 64 + i] : ((i == 64) ? Wd1[c] : 0.0f);
        Wh[idx] = (_Float16)v;
    }
    if (tid < 80) bh[tid] = (tid < 64) ? bc1[tid] : ((tid == 64) ? bd1[0] : 0.0f);
    __syncthreads();

    int m0 = wave * 16;
    int hi = lane >> 4, nn = lane & 15;
#pragma unroll
    for (int ct = 0; ct < 5; ++ct) {
        int n0 = ct * 16;
        v8f c = bias_frag(bh, n0, lane);
#pragma unroll
        for (int k0 = 0; k0 < 256; k0 += 32) {
            v16h a = load_a_frag(Fs, 256, m0, k0, lane);
            v16h b = load_b_frag(Wh, 256, n0, k0, lane);
            c = WMMA_F16(a, b, c);
        }
        if (ct < 4) {
            store_relu_frag(Hn, 64, m0, n0, lane, c);    // netc = relu(f@Wc1+bc1)
        } else if (nn == 0) {                            // col 64 == density head
#pragma unroll
            for (int r = 0; r < 8; ++r)
                out_d[v0 + m0 + r + 8 * hi] = fmaxf(c[r], 0.0f);
        }
    }
    __syncthreads();

    // c = netc @ Wc2 + bc2  (64 -> 3, tiny)
    if (tid < 128) {
        int m = tid, v = v0 + m;
#pragma unroll
        for (int a = 0; a < 3; ++a) {
            float s = bc2[a];
            for (int i = 0; i < 64; ++i)
                s += (float)Hn[m * 64 + i] * Wc2[i * 3 + a];
            out_c[v * 3 + a] = s;
        }
    }
}

// ---------------------------------------------------------------------------
// Host launcher
// ---------------------------------------------------------------------------
extern "C" void kernel_launch(void* const* d_in, const int* in_sizes, int n_in,
                              void* d_out, int out_size, void* d_ws, size_t ws_size,
                              hipStream_t stream)
{
    (void)in_sizes; (void)n_in; (void)out_size;
    const float* poses = (const float*)d_in[0];
    const float* t_ped = (const float*)d_in[1];
    const float* w     = (const float*)d_in[2];
    const float* wpts  = (const float*)d_in[3];
    const float* nodes = (const float*)d_in[4];
    const float* lc    = (const float*)d_in[5];
    const float* Wec1  = (const float*)d_in[6];  const float* bec1  = (const float*)d_in[7];
    const float* Wec21 = (const float*)d_in[8];  const float* bec21 = (const float*)d_in[9];
    const float* Wec22 = (const float*)d_in[10]; const float* bec22 = (const float*)d_in[11];
    const float* Wdc1  = (const float*)d_in[12]; const float* bdc1  = (const float*)d_in[13];
    const float* Wdc21 = (const float*)d_in[14]; const float* bdc21 = (const float*)d_in[15];
    const float* Wdc22 = (const float*)d_in[16]; const float* bdc22 = (const float*)d_in[17];
    const float* Wf1   = (const float*)d_in[18]; const float* bf1   = (const float*)d_in[19];
    const float* Wf2   = (const float*)d_in[20]; const float* bf2   = (const float*)d_in[21];
    const float* Wf3   = (const float*)d_in[22]; const float* bf3   = (const float*)d_in[23];
    const float* Wf4   = (const float*)d_in[24]; const float* bf4   = (const float*)d_in[25];
    const float* Wc1   = (const float*)d_in[26]; const float* bc1   = (const float*)d_in[27];
    const float* Wc2   = (const float*)d_in[28]; const float* bc2   = (const float*)d_in[29];
    const float* Wd1   = (const float*)d_in[30]; const float* bd1   = (const float*)d_in[31];

    // d_out layout (floats): c[6144] | d[2048] | ei[4096] | delta[384] | mean[1024] | logvar[1024]
    float* out    = (float*)d_out;
    float* o_c    = out;
    float* o_d    = out + 6144;
    float* o_ei   = out + 8192;
    float* o_dn   = out + 12288;
    float* o_mean = out + 12672;
    float* o_logv = out + 13696;

    // workspace: wgt[128][2048] f32 (1 MB) | partial[G][2048][256] f32 (G*2 MB)
    float* wgt  = (float*)d_ws;
    float* part = (float*)d_ws + 128 * 2048;

    int G = 1;
    if      (ws_size >= 1048576ull + 8ull * 2097152ull) G = 8;
    else if (ws_size >= 1048576ull + 4ull * 2097152ull) G = 4;
    else if (ws_size >= 1048576ull + 2ull * 2097152ull) G = 2;

    encdec_kernel<<<128, 64, 0, stream>>>(poses, t_ped, w,
        Wec1, bec1, Wec21, bec21, Wec22, bec22,
        Wdc1, bdc1, Wdc21, bdc21, Wdc22, bdc22,
        o_ei, o_dn, o_mean, o_logv);

    blendw_kernel<<<8, 256, 0, stream>>>(wpts, nodes, wgt);

    feat_blend_kernel<<<dim3(32, G), 256, 0, stream>>>(o_ei, lc,
        Wf1, bf1, Wf2, bf2, Wf3, bf3, Wf4, bf4,
        wgt, part, 128 / G);

    head_kernel<<<16, 256, 123264, stream>>>(part, G,
        Wc1, bc1, Wc2, bc2, Wd1, bd1, o_c, o_d);
}
